// STHP_8650064134740
// MI455X (gfx1250) — matrix operand order repaired and checked by reading, not compiled, not run on hardware
//
#include <hip/hip_runtime.h>
#include <cstdint>

// ---------------- model dims ----------------
static constexpr int T_DIM = 4, B_DIM = 8, C_INC = 2, H_IMG = 64, W_IMG = 64;
static constexpr int EMBC = 384, HEADS = 8, DHEAD = 48, DPAD = 64, NTOK = 256;
static constexpr int MTOK = T_DIM * B_DIM * NTOK;   // 8192 tokens
static constexpr int HIDC = 4 * EMBC;               // 1536
static constexpr int TBn  = T_DIM * B_DIM;          // 32
static constexpr float EPSf = 1e-5f;

typedef _Float16 h16 __attribute__((ext_vector_type(16)));
typedef _Float16 h8  __attribute__((ext_vector_type(8)));
typedef float    f8  __attribute__((ext_vector_type(8)));

// =====================================================================
// Batched WMMA GEMM:  C = alpha * A @ B^T (+ per-col scale/shift epilogue)
//   A: fp16 [M x K], row-major, ld ldA; batch offset b1*oA1 + b2*oA2
//   B: fp16 [Nc x K], row-major (row = output column, contiguous in K), ld ldB
//   C: optional fp32 and/or fp16 outputs, ldC + batch offsets
// Requirements (guaranteed by caller): M%16==0, Nc%16==0, K%32==0,
// all pointers/ld/batch-offsets 16B aligned. One wave per 16x16 tile,
// 8 waves per block along M. All fragment loads are unconditional b128s.
// =====================================================================
__global__ void gemm_wmma_k(
    const _Float16* __restrict__ A, long ldA, long oA1, long oA2,
    const _Float16* __restrict__ Bm, long ldB, long oB1, long oB2,
    float* __restrict__ Cf, _Float16* __restrict__ Ch, long ldC, long oC1, long oC2,
    int M, int Nc, int K, int nb2,
    const float* __restrict__ colScale, const float* __restrict__ colShift, float alpha)
{
  int bb = blockIdx.z;
  int b1 = bb / nb2, b2 = bb % nb2;
  int wave = threadIdx.x >> 5;
  int lane = threadIdx.x & 31;
  int mt = blockIdx.y * 8 + wave;
  if (mt * 16 >= M) return;                 // wave-uniform: EXEC all-ones at WMMA
  int row0 = mt * 16, col0 = blockIdx.x * 16;
  int rowA = row0 + (lane & 15);
  int colB = col0 + (lane & 15);
  int hi   = lane >> 4;                     // lane half selects K sub-range

  // A 16x32 in this code's contiguous k order (not AMD's table order): lanes 0-15 hold K 0..15,
  // lanes 16-31 hold K 16..31  -> two contiguous 8-half loads.
  const _Float16* ap = A + (long)b1 * oA1 + (long)b2 * oA2
                         + (long)rowA * ldA + (hi << 4);
  // B 32x16 in the same contiguous k order as A: lane = column, lanes 0-15 K 0..15, lanes 16-31 K 16..31
  // with B stored [Nc x K] -> one contiguous 16-half load.
  const _Float16* bp = Bm + (long)b1 * oB1 + (long)b2 * oB2
                         + (long)colB * ldB + (hi << 4);

  f8 acc = {0.f, 0.f, 0.f, 0.f, 0.f, 0.f, 0.f, 0.f};

  for (int k0 = 0; k0 < K; k0 += 32) {
    h8  a0 = *(const h8*)(ap + k0);
    h8  a1 = *(const h8*)(ap + k0 + 8);
    h16 b  = *(const h16*)(bp + k0);
    __builtin_prefetch(ap + k0 + 256, 0, 1);   // global_prefetch_b8 (speculative-safe)
    h16 a = __builtin_shufflevector(a0, a1, 0, 1, 2, 3, 4, 5, 6, 7,
                                    8, 9, 10, 11, 12, 13, 14, 15);
    acc = __builtin_amdgcn_wmma_f32_16x16x32_f16(false, a, false, b, (short)0, acc,
                                                 false, false);
  }

  long cbase = (long)b1 * oC1 + (long)b2 * oC2 + colB;
#pragma unroll
  for (int r = 0; r < 8; ++r) {
    int rowC = row0 + r + (hi << 3);
    float v = acc[r];
    if (colScale) v = v * colScale[colB] + colShift[colB];
    else          v = v * alpha;
    long off = cbase + (long)rowC * ldC;
    if (Cf) Cf[off] = v;
    if (Ch) Ch[off] = (_Float16)v;
  }
}

// =====================================================================
// Direct 3x3 stride-1 pad-1 conv with BN folded (eval): out = acc*sc[co]+sh[co]
// =====================================================================
template <typename Tin>
__global__ void conv3x3_bn_k(const Tin* __restrict__ in, const float* __restrict__ w,
                             const float* __restrict__ sc, const float* __restrict__ sh,
                             float* __restrict__ out, int NB, int Cin, int Cout,
                             int H, int W)
{
  long idx = (long)blockIdx.x * blockDim.x + threadIdx.x;
  long total = (long)NB * Cout * H * W;
  if (idx >= total) return;
  int x = (int)(idx % W); long t = idx / W;
  int y = (int)(t % H);   t /= H;
  int co = (int)(t % Cout);
  int n  = (int)(t / Cout);
  const Tin* inN = in + (long)n * Cin * H * W;
  const float* wC = w + (long)co * Cin * 9;
  float acc = 0.f;
  for (int ci = 0; ci < Cin; ++ci) {
    const Tin* ip = inN + (long)ci * H * W;
    const float* wp = wC + ci * 9;
#pragma unroll
    for (int ky = 0; ky < 3; ++ky) {
      int yy = y + ky - 1;
      if ((unsigned)yy >= (unsigned)H) continue;
#pragma unroll
      for (int kx = 0; kx < 3; ++kx) {
        int xx = x + kx - 1;
        if ((unsigned)xx >= (unsigned)W) continue;
        acc += (float)ip[(long)yy * W + xx] * wp[ky * 3 + kx];
      }
    }
  }
  out[idx] = acc * sc[co] + sh[co];
}

// MaxPool2d(kernel=3, stride=2, padding=1): H,W even -> H/2, W/2
__global__ void maxpool3s2_k(const float* __restrict__ in, float* __restrict__ out,
                             int NC, int H, int W)
{
  int Ho = H / 2, Wo = W / 2;
  long idx = (long)blockIdx.x * blockDim.x + threadIdx.x;
  long total = (long)NC * Ho * Wo;
  if (idx >= total) return;
  int xo = (int)(idx % Wo); long t = idx / Wo;
  int yo = (int)(t % Ho);
  int nc = (int)(t / Ho);
  const float* ip = in + (long)nc * H * W;
  float m = -3.4e38f;
#pragma unroll
  for (int ky = 0; ky < 3; ++ky) {
    int yy = 2 * yo + ky - 1;
    if ((unsigned)yy >= (unsigned)H) continue;
#pragma unroll
    for (int kx = 0; kx < 3; ++kx) {
      int xx = 2 * xo + kx - 1;
      if ((unsigned)xx >= (unsigned)W) continue;
      float v = ip[(long)yy * W + xx];
      m = v > m ? v : m;
    }
  }
  out[idx] = m;
}

// Multi-step LIF over leading T axis (tau=2, decay_input, hard reset, v_reset=0)
__global__ void lif_k(const float* __restrict__ x, _Float16* __restrict__ s,
                      long strideT, int T, float vth)
{
  long i = (long)blockIdx.x * blockDim.x + threadIdx.x;
  if (i >= strideT) return;
  float v = 0.f;
  for (int t = 0; t < T; ++t) {
    float xt = x[(long)t * strideT + i];
    v += (xt - v) * 0.5f;
    float sp = (v >= vth) ? 1.f : 0.f;
    s[(long)t * strideT + i] = (_Float16)sp;
    v *= (1.f - sp);
  }
}

// LIF that scatters q/k spikes into head-major zero-padded layout:
//   out[((t*B+b)*HEADS+hh)][n][DPAD], dd<DHEAD written (pad pre-zeroed once)
__global__ void lif_qkpad_k(const float* __restrict__ x, _Float16* __restrict__ out,
                            float vth)
{
  const long strideT = (long)B_DIM * NTOK * EMBC;
  long i = (long)blockIdx.x * blockDim.x + threadIdx.x;
  if (i >= strideT) return;
  int c  = (int)(i % EMBC);
  int nn = (int)((i / EMBC) % NTOK);
  int b  = (int)(i / ((long)EMBC * NTOK));
  int hh = c / DHEAD, dd = c % DHEAD;
  float v = 0.f;
  for (int t = 0; t < T_DIM; ++t) {
    float xt = x[(long)t * strideT + i];
    v += (xt - v) * 0.5f;
    float sp = (v >= vth) ? 1.f : 0.f;
    long o = (((long)(t * B_DIM + b) * HEADS + hh) * NTOK + nn) * DPAD + dd;
    out[o] = (_Float16)sp;
    v *= (1.f - sp);
  }
}

// LIF that emits v spikes transposed per head: out[((t*B+b)*HEADS+hh)][dd][n]
__global__ void lif_vt_k(const float* __restrict__ x, _Float16* __restrict__ out,
                         float vth)
{
  const long strideT = (long)B_DIM * NTOK * EMBC;
  long i = (long)blockIdx.x * blockDim.x + threadIdx.x;
  if (i >= strideT) return;
  int c  = (int)(i % EMBC);
  int nn = (int)((i / EMBC) % NTOK);
  int b  = (int)(i / ((long)EMBC * NTOK));
  int hh = c / DHEAD, dd = c % DHEAD;
  float v = 0.f;
  for (int t = 0; t < T_DIM; ++t) {
    float xt = x[(long)t * strideT + i];
    v += (xt - v) * 0.5f;
    float sp = (v >= vth) ? 1.f : 0.f;
    long o = ((long)(t * B_DIM + b) * HEADS + hh) * DHEAD * NTOK + (long)dd * NTOK + nn;
    out[o] = (_Float16)sp;
    v *= (1.f - sp);
  }
}

// fold BN (+optional linear bias) into per-channel scale/shift
__global__ void fold_bn_k(const float* __restrict__ g, const float* __restrict__ b,
                          const float* __restrict__ m, const float* __restrict__ v,
                          const float* __restrict__ bias, float* __restrict__ sc,
                          float* __restrict__ sh, int C)
{
  int i = blockIdx.x * blockDim.x + threadIdx.x;
  if (i >= C) return;
  float s = g[i] * rsqrtf(v[i] + EPSf);
  sc[i] = s;
  sh[i] = b[i] - m[i] * s + (bias ? bias[i] * s : 0.f);
}

__global__ void f2h_k(const float* __restrict__ a, _Float16* __restrict__ o, long n)
{
  long i = (long)blockIdx.x * blockDim.x + threadIdx.x;
  if (i < n) o[i] = (_Float16)a[i];
}
__global__ void h2f_k(const _Float16* __restrict__ a, float* __restrict__ o, long n)
{
  long i = (long)blockIdx.x * blockDim.x + threadIdx.x;
  if (i < n) o[i] = (float)a[i];
}
__global__ void zeroh_k(_Float16* __restrict__ p, long n)
{
  long i = (long)blockIdx.x * blockDim.x + threadIdx.x;
  if (i < n) p[i] = (_Float16)0.f;
}
__global__ void addh_k(_Float16* __restrict__ y, const _Float16* __restrict__ s, long n)
{
  long i = (long)blockIdx.x * blockDim.x + threadIdx.x;
  if (i < n) y[i] = (_Float16)((float)y[i] + (float)s[i]);
}
// tok[tb,n,c] layout <-> img[tb,c,n] layout
__global__ void tok2img_k(const _Float16* __restrict__ tok, _Float16* __restrict__ img,
                          int NBt, int N, int C)
{
  long i = (long)blockIdx.x * blockDim.x + threadIdx.x;
  long total = (long)NBt * N * C;
  if (i >= total) return;
  int c = (int)(i % C); long t = i / C;
  int n = (int)(t % N);
  int tb = (int)(t / N);
  img[(long)tb * C * N + (long)c * N + n] = tok[i];
}
__global__ void img2tok_add_k(_Float16* __restrict__ tok, const _Float16* __restrict__ img,
                              int NBt, int N, int C)
{
  long i = (long)blockIdx.x * blockDim.x + threadIdx.x;
  long total = (long)NBt * N * C;
  if (i >= total) return;
  int c = (int)(i % C); long t = i / C;
  int n = (int)(t % N);
  int tb = (int)(t / N);
  float v = (float)tok[i] + (float)img[(long)tb * C * N + (long)c * N + n];
  tok[i] = (_Float16)v;
}
__global__ void add2_tok_k(const _Float16* __restrict__ aimg, const _Float16* __restrict__ bimg,
                           _Float16* __restrict__ tok, int NBt, int N, int C)
{
  long i = (long)blockIdx.x * blockDim.x + threadIdx.x;
  long total = (long)NBt * N * C;
  if (i >= total) return;
  int c = (int)(i % C); long t = i / C;
  int n = (int)(t % N);
  int tb = (int)(t / N);
  long im = (long)tb * C * N + (long)c * N + n;
  tok[i] = (_Float16)((float)aimg[im] + (float)bimg[im]);
}

// ===================== host side =====================
static inline dim3 g1(long n) { return dim3((unsigned)((n + 255) / 256)); }

struct BnP { const float *b, *g, *m, *v; };
struct BlkP {
  const float *kW, *kb; BnP kbn;
  const float *pW, *pb; BnP pbn;
  const float *qW, *qb; BnP qbn;
  const float *vW, *vb; BnP vbn;
  const float *b1, *b2; BnP bn1, bn2;
  const float *rpe;     BnP rpebn;
  const float *w1, *w2;
};
struct CmlP {
  BnP bn1, bn2, bn3, bn4;
  const float *c1, *c2, *c3, *c4, *rpe;
  BnP rpebn;
};

static void gemm(hipStream_t st,
                 const _Float16* A, long ldA, long oA1, long oA2,
                 const _Float16* Bm, long ldB, long oB1, long oB2,
                 float* Cf, _Float16* Ch, long ldC, long oC1, long oC2,
                 int M, int Nc, int K, int nb1, int nb2,
                 const float* cs, const float* sh, float alpha)
{
  dim3 grid((Nc + 15) / 16, (M + 127) / 128, nb1 * nb2);
  gemm_wmma_k<<<grid, 256, 0, st>>>(A, ldA, oA1, oA2, Bm, ldB, oB1, oB2,
                                    Cf, Ch, ldC, oC1, oC2, M, Nc, K, nb2, cs, sh, alpha);
}

extern "C" void kernel_launch(void* const* d_in, const int* in_sizes, int n_in,
                              void* d_out, int out_size, void* d_ws, size_t ws_size,
                              hipStream_t stream)
{
  // ---- input order: 'x' then params leaves in jax-sorted-key order.
  // Hedge: if in_sizes[0] is not x's size, assume params-first / x-last.
  const long xElems = (long)T_DIM * B_DIM * C_INC * H_IMG * W_IMG; // 262144
  const float* xin;
  int p;
  if (in_sizes[0] == (int)xElems) { xin = (const float*)d_in[0]; p = 1; }
  else                            { xin = (const float*)d_in[n_in - 1]; p = 0; }

  auto nf  = [&]() -> const float* { return (const float*)d_in[p++]; };
  auto nbn = [&]() { BnP r; r.b = nf(); r.g = nf(); r.m = nf(); r.v = nf(); return r; };

  BlkP blk[4];
  for (int i = 0; i < 4; ++i) {        // block dict: 'attn' < 'mlp'
    blk[i].kW = nf(); blk[i].kb = nf(); blk[i].kbn = nbn();
    blk[i].pW = nf(); blk[i].pb = nf(); blk[i].pbn = nbn();
    blk[i].qW = nf(); blk[i].qb = nf(); blk[i].qbn = nbn();
    blk[i].vW = nf(); blk[i].vb = nf(); blk[i].vbn = nbn();
    blk[i].b1 = nf(); blk[i].b2 = nf();
    blk[i].bn1 = nbn(); blk[i].bn2 = nbn();
    blk[i].rpe = nf(); blk[i].rpebn = nbn();
    blk[i].w1 = nf(); blk[i].w2 = nf();
  }
  CmlP cml;                            // cml keys sorted: bn1..bn4, c1..c4, rpe, rpe_bn
  cml.bn1 = nbn(); cml.bn2 = nbn(); cml.bn3 = nbn(); cml.bn4 = nbn();
  cml.c1 = nf(); cml.c2 = nf(); cml.c3 = nf(); cml.c4 = nf();
  cml.rpe = nf(); cml.rpebn = nbn();

  // ---- workspace bump allocator (~300 MB total) ----
  char* wsp = (char*)d_ws;
  size_t off = 0;
  auto alloc = [&](size_t bytes) -> void* {
    void* r = wsp + off;
    off += (bytes + 255) & ~(size_t)255;
    return r;
  };
  float*    P0 = (float*)alloc((size_t)TBn * 192 * 4096 * 4);  // 100.7 MB fp32 scratch
  float*    P1 = (float*)alloc((size_t)TBn * 384 * 1024 * 4);  //  50.3 MB fp32 scratch
  _Float16* S0 = (_Float16*)alloc((size_t)MTOK * HIDC * 2);    //  25.2 MB fp16 spikes
  _Float16* S1 = (_Float16*)alloc((size_t)MTOK * HIDC * 2);
  _Float16* XF = (_Float16*)alloc((size_t)MTOK * EMBC * 2);    // x_feat spikes (img layout)
  _Float16* Y  = (_Float16*)alloc((size_t)MTOK * EMBC * 2);    // residual stream (tok layout)
  _Float16* YI = (_Float16*)alloc((size_t)MTOK * EMBC * 2);    // img-layout temp
  const long QPsz = (long)TBn * HEADS * NTOK * DPAD;           // padded q/k: 4.19M halves
  const long VTsz = (long)TBn * HEADS * DHEAD * NTOK;          // transposed v
  _Float16* Qp = (_Float16*)alloc((size_t)QPsz * 2);
  _Float16* Kp = (_Float16*)alloc((size_t)QPsz * 2);
  _Float16* Vt = (_Float16*)alloc((size_t)VTsz * 2);
  _Float16* AT = (_Float16*)alloc((size_t)TBn * HEADS * NTOK * NTOK * 2); // attn 33.5 MB
  (void)ws_size;

  auto allocF = [&](int n) { return (float*)alloc((size_t)n * 4); };
  auto allocH = [&](long n) { return (_Float16*)alloc((size_t)n * 2); };

  // ---- fold all BNs, convert all GEMM weights to fp16 ----
  auto fold = [&](BnP bn, const float* bias, int C, float** sc, float** sh) {
    *sc = allocF(C); *sh = allocF(C);
    fold_bn_k<<<g1(C), 256, 0, stream>>>(bn.g, bn.b, bn.m, bn.v, bias, *sc, *sh, C);
  };
  auto toH = [&](const float* w, long n) {
    _Float16* o = allocH(n);
    f2h_k<<<g1(n), 256, 0, stream>>>(w, o, n);
    return o;
  };

  float *cs1, *ch1, *cs2, *ch2, *cs3, *ch3, *cs4, *ch4, *csr, *chr;
  fold(cml.bn1, nullptr, 48,  &cs1, &ch1);
  fold(cml.bn2, nullptr, 96,  &cs2, &ch2);
  fold(cml.bn3, nullptr, 192, &cs3, &ch3);
  fold(cml.bn4, nullptr, 384, &cs4, &ch4);
  fold(cml.rpebn, nullptr, 384, &csr, &chr);

  _Float16 *qWh[4], *kWh[4], *vWh[4], *pWh[4], *w1h[4], *w2h[4];
  float *qs[4], *qsh[4], *ks[4], *ksh[4], *vs[4], *vsh[4], *ps[4], *psh[4];
  float *m1s[4], *m1sh[4], *m2s[4], *m2sh[4], *rs[4], *rsh[4];
  for (int i = 0; i < 4; ++i) {
    qWh[i] = toH(blk[i].qW, (long)EMBC * EMBC);
    kWh[i] = toH(blk[i].kW, (long)EMBC * EMBC);
    vWh[i] = toH(blk[i].vW, (long)EMBC * EMBC);
    pWh[i] = toH(blk[i].pW, (long)EMBC * EMBC);
    w1h[i] = toH(blk[i].w1, (long)HIDC * EMBC);
    w2h[i] = toH(blk[i].w2, (long)EMBC * HIDC);
    fold(blk[i].qbn, blk[i].qb, EMBC, &qs[i], &qsh[i]);
    fold(blk[i].kbn, blk[i].kb, EMBC, &ks[i], &ksh[i]);
    fold(blk[i].vbn, blk[i].vb, EMBC, &vs[i], &vsh[i]);
    fold(blk[i].pbn, blk[i].pb, EMBC, &ps[i], &psh[i]);
    fold(blk[i].bn1, blk[i].b1, HIDC, &m1s[i], &m1sh[i]);
    fold(blk[i].bn2, blk[i].b2, EMBC, &m2s[i], &m2sh[i]);
    fold(blk[i].rpebn, nullptr, EMBC, &rs[i], &rsh[i]);
  }

  // zero the padded q/k buffers once (LIF writes only dd<48; pad stays 0 forever)
  zeroh_k<<<g1(QPsz), 256, 0, stream>>>(Qp, QPsz);
  zeroh_k<<<g1(QPsz), 256, 0, stream>>>(Kp, QPsz);

  // ================= CML conv stem =================
  long n;
  n = (long)TBn * 48 * 4096;
  conv3x3_bn_k<float><<<g1(n), 256, 0, stream>>>(xin, cml.c1, cs1, ch1, P1, TBn, C_INC, 48, 64, 64);
  lif_k<<<g1(n / T_DIM), 256, 0, stream>>>(P1, S0, n / T_DIM, T_DIM, 1.f);

  n = (long)TBn * 96 * 4096;
  conv3x3_bn_k<_Float16><<<g1(n), 256, 0, stream>>>(S0, cml.c2, cs2, ch2, P0, TBn, 48, 96, 64, 64);
  lif_k<<<g1(n / T_DIM), 256, 0, stream>>>(P0, S1, n / T_DIM, T_DIM, 1.f);

  n = (long)TBn * 192 * 4096;
  conv3x3_bn_k<_Float16><<<g1(n), 256, 0, stream>>>(S1, cml.c3, cs3, ch3, P0, TBn, 96, 192, 64, 64);
  n = (long)TBn * 192 * 1024;
  maxpool3s2_k<<<g1(n), 256, 0, stream>>>(P0, P1, TBn * 192, 64, 64);
  lif_k<<<g1(n / T_DIM), 256, 0, stream>>>(P1, S0, n / T_DIM, T_DIM, 1.f);

  n = (long)TBn * 384 * 1024;
  conv3x3_bn_k<_Float16><<<g1(n), 256, 0, stream>>>(S0, cml.c4, cs4, ch4, P0, TBn, 192, 384, 32, 32);
  n = (long)TBn * 384 * 256;
  maxpool3s2_k<<<g1(n), 256, 0, stream>>>(P0, P1, TBn * 384, 32, 32);
  lif_k<<<g1(n / T_DIM), 256, 0, stream>>>(P1, XF, n / T_DIM, T_DIM, 1.f);   // x_feat spikes

  conv3x3_bn_k<_Float16><<<g1(n), 256, 0, stream>>>(XF, cml.rpe, csr, chr, P1, TBn, 384, 384, 16, 16);
  lif_k<<<g1(n / T_DIM), 256, 0, stream>>>(P1, S0, n / T_DIM, T_DIM, 1.f);
  add2_tok_k<<<g1(n), 256, 0, stream>>>(S0, XF, Y, TBn, NTOK, EMBC);         // tokens (T,B,N,C)

  // ================= transformer blocks =================
  const long strideTok = (long)B_DIM * NTOK * EMBC;   // LIF T-stride on (T,B,N,C)
  const long nTok = (long)MTOK * EMBC;
  for (int i = 0; i < 4; ++i) {
    // ---- SSA: q/k/v linears + BN -> LIF spikes (q/k head-padded, v head-transposed)
    gemm(stream, Y, EMBC, 0, 0, qWh[i], EMBC, 0, 0, P0, nullptr, EMBC, 0, 0,
         MTOK, EMBC, EMBC, 1, 1, qs[i], qsh[i], 1.f);
    lif_qkpad_k<<<g1(strideTok), 256, 0, stream>>>(P0, Qp, 1.f);
    gemm(stream, Y, EMBC, 0, 0, kWh[i], EMBC, 0, 0, P0, nullptr, EMBC, 0, 0,
         MTOK, EMBC, EMBC, 1, 1, ks[i], ksh[i], 1.f);
    lif_qkpad_k<<<g1(strideTok), 256, 0, stream>>>(P0, Kp, 1.f);
    gemm(stream, Y, EMBC, 0, 0, vWh[i], EMBC, 0, 0, P0, nullptr, EMBC, 0, 0,
         MTOK, EMBC, EMBC, 1, 1, vs[i], vsh[i], 1.f);
    lif_vt_k<<<g1(strideTok), 256, 0, stream>>>(P0, Vt, 1.f);

    // ---- attn = (q @ k^T) * 0.125, batched over T*B*heads; K padded 48->64
    gemm(stream,
         Qp, DPAD, (long)NTOK * DPAD, 0,
         Kp, DPAD, (long)NTOK * DPAD, 0,
         nullptr, AT, NTOK, (long)NTOK * NTOK, 0,
         NTOK, NTOK, DPAD, TBn * HEADS, 1, nullptr, nullptr, 0.125f);
    // ---- o = attn @ v  (B = v^T per head: [d x N], contiguous in K)
    gemm(stream,
         AT, NTOK, (long)HEADS * NTOK * NTOK, (long)NTOK * NTOK,
         Vt, NTOK, (long)HEADS * DHEAD * NTOK, (long)DHEAD * NTOK,
         P0, nullptr, EMBC, (long)NTOK * EMBC, DHEAD,
         NTOK, DHEAD, NTOK, TBn, HEADS, nullptr, nullptr, 1.f);
    lif_k<<<g1(strideTok), 256, 0, stream>>>(P0, S0, strideTok, T_DIM, 0.5f);  // v_th=0.5
    // ---- proj + BN -> LIF -> residual add
    gemm(stream, S0, EMBC, 0, 0, pWh[i], EMBC, 0, 0, P0, nullptr, EMBC, 0, 0,
         MTOK, EMBC, EMBC, 1, 1, ps[i], psh[i], 1.f);
    lif_k<<<g1(strideTok), 256, 0, stream>>>(P0, S1, strideTok, T_DIM, 1.f);
    addh_k<<<g1(nTok), 256, 0, stream>>>(Y, S1, nTok);

    // ---- MLP: fc1 -> LIF -> fc2 -> LIF -> rpe conv -> LIF -> residual add
    gemm(stream, Y, EMBC, 0, 0, w1h[i], EMBC, 0, 0, P0, nullptr, HIDC, 0, 0,
         MTOK, HIDC, EMBC, 1, 1, m1s[i], m1sh[i], 1.f);
    long strideHid = (long)B_DIM * NTOK * HIDC;
    lif_k<<<g1(strideHid), 256, 0, stream>>>(P0, S0, strideHid, T_DIM, 1.f);
    gemm(stream, S0, HIDC, 0, 0, w2h[i], HIDC, 0, 0, P1, nullptr, EMBC, 0, 0,
         MTOK, EMBC, HIDC, 1, 1, m2s[i], m2sh[i], 1.f);
    lif_k<<<g1(strideTok), 256, 0, stream>>>(P1, S1, strideTok, T_DIM, 1.f);
    tok2img_k<<<g1(nTok), 256, 0, stream>>>(S1, YI, TBn, NTOK, EMBC);
    conv3x3_bn_k<_Float16><<<g1(nTok), 256, 0, stream>>>(YI, blk[i].rpe, rs[i], rsh[i],
                                                         P1, TBn, EMBC, EMBC, 16, 16);
    lif_k<<<g1(strideTok), 256, 0, stream>>>(P1, S0, strideTok, T_DIM, 1.f);
    img2tok_add_k<<<g1(nTok), 256, 0, stream>>>(Y, S0, TBn, NTOK, EMBC);
  }

  // ---- final output: (T,B,N,C) fp32
  h2f_k<<<g1(nTok), 256, 0, stream>>>(Y, (float*)d_out, nTok);
  (void)out_size;
}
